// RelPosAttention_13254269075863
// MI455X (gfx1250) — matrix operand order repaired and checked by reading, not compiled, not run on hardware
//
#include <hip/hip_runtime.h>

// ---------- types ----------
typedef __attribute__((ext_vector_type(16))) __bf16 bf16x16;
typedef __attribute__((ext_vector_type(8)))  __bf16 bf16x8;
typedef __attribute__((ext_vector_type(8)))  float  f32x8;
typedef __attribute__((ext_vector_type(4)))  int    i32x4;

// problem constants: N=8, H=W=32, C=768, heads=12, hd=64, HW=1024, B=96
#define NIMG 8
#define CDIM 768
#define NHEADS 12
#define HD 64
#define HWDIM 1024
#define BATT 96   // NIMG * NHEADS
#define MROWS 8192

#ifndef __has_builtin
#define __has_builtin(x) 0
#endif

#if __has_builtin(__builtin_amdgcn_global_load_async_to_lds_b128)
#define HAVE_ASYNC_LDS 1
#else
#define HAVE_ASYNC_LDS 0
#endif

// pointer-to-AS-qualified-vector helper types
typedef __attribute__((address_space(1))) i32x4 gmem_i32x4;
typedef __attribute__((address_space(3))) i32x4 lds_i32x4;

// 16-byte global -> LDS copy: async (ASYNCcnt-tracked) when available.
__device__ __forceinline__ void async_copy16(void* lds_dst, const void* gsrc) {
#if HAVE_ASYNC_LDS
  __builtin_amdgcn_global_load_async_to_lds_b128(
      (gmem_i32x4*)gsrc, (lds_i32x4*)lds_dst, 0, 0);
#else
  *reinterpret_cast<uint4*>(lds_dst) = *reinterpret_cast<const uint4*>(gsrc);
#endif
}

__device__ __forceinline__ void wait_async0() {
#if __has_builtin(__builtin_amdgcn_s_wait_asynccnt)
  __builtin_amdgcn_s_wait_asynccnt(0);
#elif HAVE_ASYNC_LDS
  asm volatile("s_wait_asynccnt 0" ::: "memory");
#endif
}

__device__ __forceinline__ bf16x16 join8(bf16x8 lo, bf16x8 hi) {
  return __builtin_shufflevector(lo, hi, 0,1,2,3,4,5,6,7,8,9,10,11,12,13,14,15);
}

// A-fragment (16x32 bf16), tile origin `base`, row stride `stride` (elements).
// ISA 7.12.2: lane row = lane&15 ; lanes 0-15 hold K 0..7 & 16..23, lanes 16-31 hold K 8..15 & 24..31
__device__ __forceinline__ bf16x16 load_frag_a(const __bf16* base, int stride, int lane) {
  const int row = lane & 15;
  const int kb  = (lane >> 4) << 3;                // 0 or 8
  const __bf16* p = base + row * stride;
  bf16x8 lo = *reinterpret_cast<const bf16x8*>(p + kb);
  bf16x8 hi = *reinterpret_cast<const bf16x8*>(p + 16 + kb);
  return join8(lo, hi);
}

// B-fragment (32x16 bf16) from tile stored column-major: sT[n * stride + k]
// ISA: lanes 0-15 -> col=lane, K 0..7 & 16..23 ; lanes 16-31 -> col=lane-16, K 8..15 & 24..31
__device__ __forceinline__ bf16x16 load_frag_b(const __bf16* base, int stride, int lane) {
  const int col = lane & 15;
  const int kb  = (lane >> 4) << 3;                // 0 or 8
  const __bf16* p = base + col * stride + kb;
  bf16x8 lo = *reinterpret_cast<const bf16x8*>(p);
  bf16x8 hi = *reinterpret_cast<const bf16x8*>(p + 16);
  return join8(lo, hi);
}

// ------------------------------------------------------------------
// Kernel 1: qkv = x(8192x768) @ w_qkv(768x2304) + b_qkv
// q,k: head-major bf16 (96, 1024, 64); v: transposed bf16 (96, 64, 1024)
// ------------------------------------------------------------------
__global__ __launch_bounds__(256)
void qkv_gemm_kernel(const float* __restrict__ x, const float* __restrict__ wqkv,
                     const float* __restrict__ bqkv,
                     __bf16* __restrict__ qb, __bf16* __restrict__ kb, __bf16* __restrict__ vbT)
{
  __shared__ __align__(16) __bf16 sA[128 * 32];       // (m,k)
  __shared__ __align__(16) __bf16 sB[128 * 32];       // (n,k)  (transposed stage)
  const int mbase = blockIdx.y * 128;
  const int nbase = blockIdx.x * 128;
  const int t = threadIdx.x;
  const int lane = t & 31, w = t >> 5;
  const int warpM = w & 3, warpN = w >> 2;            // 4x2 waves -> 32x64 per wave

  const f32x8 fzero = {};
  f32x8 acc[2][4];
  #pragma unroll
  for (int i = 0; i < 2; ++i)
    #pragma unroll
    for (int j = 0; j < 4; ++j) acc[i][j] = fzero;

  for (int kk = 0; kk < CDIM; kk += 32) {
    // ---- batch all staging loads first (one clause, one wait) ----
    float4 va[4], vbw[4];
    #pragma unroll
    for (int j = 0; j < 4; ++j) {
      const int f = t + 256 * j;
      const int row = f >> 3, c4 = (f & 7) << 2;
      va[j] = *reinterpret_cast<const float4*>(
          &x[(size_t)(mbase + row) * CDIM + kk + c4]);
    }
    #pragma unroll
    for (int j = 0; j < 4; ++j) {
      const int f = t + 256 * j;
      const int k = f >> 5, n4 = (f & 31) << 2;
      vbw[j] = *reinterpret_cast<const float4*>(
          &wqkv[(size_t)(kk + k) * 2304 + nbase + n4]);
    }
    // ---- then convert + store to LDS ----
    #pragma unroll
    for (int j = 0; j < 4; ++j) {
      const int f = t + 256 * j;
      const int row = f >> 3, c4 = (f & 7) << 2;
      __bf16* d = &sA[row * 32 + c4];
      d[0] = (__bf16)va[j].x; d[1] = (__bf16)va[j].y;
      d[2] = (__bf16)va[j].z; d[3] = (__bf16)va[j].w;
    }
    #pragma unroll
    for (int j = 0; j < 4; ++j) {
      const int f = t + 256 * j;
      const int k = f >> 5, n4 = (f & 31) << 2;
      sB[(n4 + 0) * 32 + k] = (__bf16)vbw[j].x;
      sB[(n4 + 1) * 32 + k] = (__bf16)vbw[j].y;
      sB[(n4 + 2) * 32 + k] = (__bf16)vbw[j].z;
      sB[(n4 + 3) * 32 + k] = (__bf16)vbw[j].w;
    }
    if (kk + 32 < CDIM) {
      __builtin_prefetch(&x[(size_t)(mbase + (t >> 1)) * CDIM + kk + 32], 0, 3);
      __builtin_prefetch(&wqkv[(size_t)(kk + 32 + (t >> 3)) * 2304 + nbase], 0, 3);
    }
    __syncthreads();

    bf16x16 fa[2], fb[4];
    #pragma unroll
    for (int mt = 0; mt < 2; ++mt)
      fa[mt] = load_frag_a(&sA[(warpM * 32 + mt * 16) * 32], 32, lane);
    #pragma unroll
    for (int nt = 0; nt < 4; ++nt)
      fb[nt] = load_frag_b(&sB[(warpN * 64 + nt * 16) * 32], 32, lane);
    #pragma unroll
    for (int mt = 0; mt < 2; ++mt)
      #pragma unroll
      for (int nt = 0; nt < 4; ++nt)
        acc[mt][nt] = __builtin_amdgcn_wmma_f32_16x16x32_bf16(
            false, fa[mt], false, fb[nt], (short)0, acc[mt][nt], false, false);
    __syncthreads();
  }

  // epilogue: +bias, scatter into q/k head-major + v transposed
  const int M0 = (lane >> 4) << 3;
  const int nc = lane & 15;
  #pragma unroll
  for (int mt = 0; mt < 2; ++mt)
    #pragma unroll
    for (int nt = 0; nt < 4; ++nt) {
      const int nn = nbase + warpN * 64 + nt * 16 + nc;     // 0..2303
      const float bias = bqkv[nn];
      const int which = nn / CDIM;
      const int cc = nn - which * CDIM;
      const int head = cc >> 6, dch = cc & 63;
      #pragma unroll
      for (int r = 0; r < 8; ++r) {
        const int mm = mbase + warpM * 32 + mt * 16 + M0 + r;
        const int nimg = mm >> 10, hw = mm & 1023;
        const size_t bh_ = (size_t)(nimg * NHEADS + head);
        const float val = acc[mt][nt][r] + bias;
        if (which == 0)      qb[(bh_ * HWDIM + hw) * HD + dch] = (__bf16)val;
        else if (which == 1) kb[(bh_ * HWDIM + hw) * HD + dch] = (__bf16)val;
        else                 vbT[(bh_ * HD + dch) * HWDIM + hw] = (__bf16)val;
      }
    }
}

// ------------------------------------------------------------------
// Kernel 2: decomposed rel-pos bias rows
// bias_h[b,qi,kj] = sum_d q[b,qi,d] * rel_pos_h[(qi>>5)-kj+31, d]   (and _w with qi&31)
// ------------------------------------------------------------------
__global__ __launch_bounds__(256)
void relpos_bias_kernel(const __bf16* __restrict__ qb,
                        const float* __restrict__ rph, const float* __restrict__ rpw,
                        float* __restrict__ bh, float* __restrict__ bw)
{
  int idx = blockIdx.x * 256 + threadIdx.x;           // 2 * 96 * 1024 * 32 total
  const int which = idx / (BATT * HWDIM * 32);
  idx -= which * (BATT * HWDIM * 32);
  const int b  = idx >> 15;                           // /(1024*32)
  const int qi = (idx >> 5) & 1023;
  const int kj = idx & 31;

  const __bf16* q = &qb[((size_t)b * HWDIM + qi) * HD];
  const float* tab;
  if (which == 0) { const int h = qi >> 5; tab = &rph[(h  - kj + 31) * HD]; }
  else            { const int ww = qi & 31; tab = &rpw[(ww - kj + 31) * HD]; }

  float s = 0.f;
  #pragma unroll
  for (int d = 0; d < HD; ++d) s += (float)q[d] * tab[d];

  float* out = (which == 0) ? bh : bw;
  out[((size_t)b * HWDIM + qi) * 32 + kj] = s;
}

// ------------------------------------------------------------------
// Kernel 3: flash attention with decomposed rel-pos bias.
// block = (batch b, q-tile of 128 rows); 8 waves x 16 q-rows; 32-key steps,
// double-buffered async K/V staging.
// ------------------------------------------------------------------
__global__ __launch_bounds__(256)
void attn_kernel(const __bf16* __restrict__ qb, const __bf16* __restrict__ kb,
                 const __bf16* __restrict__ vbT,
                 const float* __restrict__ bh, const float* __restrict__ bw,
                 float* __restrict__ ob)
{
  __shared__ __align__(16) __bf16 sQ[128 * 64];      // (qrow, d)
  __shared__ __align__(16) __bf16 sK[2][32 * 64];    // (key, d)   == B-layout for QK^T
  __shared__ __align__(16) __bf16 sVT[2][64 * 32];   // (d, key)   == B-layout for P V
  __shared__ __align__(16) float  sBH[128 * 32];
  __shared__ __align__(16) float  sBW[128 * 32];
  __shared__ __align__(16) __bf16 sP[8][16 * 32];    // per-wave P tile

  const int b = blockIdx.x;           // 0..95
  const int qbase = blockIdx.y * 128; // 8 q-tiles
  const int t = threadIdx.x, lane = t & 31, w = t >> 5;
  const int M0 = (lane >> 4) << 3;    // row half offset
  const int nc = lane & 15;           // column-in-tile

  // K tile: 32x64 bf16 = 256 x 16B, one copy per thread
  // VT tile: 64 rows of 32 keys; thread t -> row d = t>>2, 16B segment (t&3)
  const int vt_d = t >> 2, vt_seg = (t & 3) << 3;
  auto stage_kv = [&](int buf, int kb0) {
    async_copy16(&sK[buf][t * 8], &kb[((size_t)b * HWDIM + kb0) * HD + t * 8]);
    async_copy16(&sVT[buf][vt_d * 32 + vt_seg],
                 &vbT[((size_t)b * HD + vt_d) * HWDIM + kb0 + vt_seg]);
  };

  // prologue: async-stage Q tile, bias rows, first K/V tile
  #pragma unroll
  for (int j = 0; j < 4; ++j) {
    const int e = t + 256 * j;
    async_copy16(&sQ[e * 8], &qb[((size_t)b * HWDIM + qbase) * HD + e * 8]);
    async_copy16(&sBH[e * 4], &bh[((size_t)b * HWDIM + qbase) * 32 + e * 4]);
    async_copy16(&sBW[e * 4], &bw[((size_t)b * HWDIM + qbase) * 32 + e * 4]);
  }
  stage_kv(0, 0);
  wait_async0();
  __syncthreads();

  const f32x8 fzero = {};
  float m_i[8], l_i[8];
  #pragma unroll
  for (int r = 0; r < 8; ++r) { m_i[r] = -3.0e38f; l_i[r] = 0.f; }
  f32x8 o[4];
  #pragma unroll
  for (int n = 0; n < 4; ++n) o[n] = fzero;

  int cur = 0;
  for (int kt = 0; kt < 32; ++kt) {
    const int kbase = kt * 32;
    if (kt + 1 < 32) stage_kv(cur ^ 1, kbase + 32);   // overlap with compute below

    // S = Q K^T  (per wave: 16 x 32 scores, contraction hd=64 in two chunks)
    f32x8 s[2]; s[0] = fzero; s[1] = fzero;
    #pragma unroll
    for (int c0 = 0; c0 < 64; c0 += 32) {
      const bf16x16 fa = load_frag_a(&sQ[(w * 16) * 64 + c0], 64, lane);
      #pragma unroll
      for (int nt = 0; nt < 2; ++nt) {
        const bf16x16 fb = load_frag_b(&sK[cur][(nt * 16) * 64 + c0], 64, lane);
        s[nt] = __builtin_amdgcn_wmma_f32_16x16x32_bf16(
            false, fa, false, fb, (short)0, s[nt], false, false);
      }
    }

    // scale + decomposed bias
    #pragma unroll
    for (int nt = 0; nt < 2; ++nt)
      #pragma unroll
      for (int r = 0; r < 8; ++r) {
        const int qrl = w * 16 + M0 + r;
        const int key = kbase + nt * 16 + nc;
        s[nt][r] = s[nt][r] * 0.125f + sBH[qrl * 32 + (key >> 5)] + sBW[qrl * 32 + (key & 31)];
      }

    // online softmax: rows live across 16-lane halves
    float pscale[8];
    #pragma unroll
    for (int r = 0; r < 8; ++r) {
      float mx = fmaxf(s[0][r], s[1][r]);
      #pragma unroll
      for (int off = 1; off < 16; off <<= 1) mx = fmaxf(mx, __shfl_xor(mx, off, 32));
      const float mnew = fmaxf(m_i[r], mx);
      const float sc = __expf(m_i[r] - mnew);
      const float p0 = __expf(s[0][r] - mnew);
      const float p1 = __expf(s[1][r] - mnew);
      s[0][r] = p0; s[1][r] = p1;
      float ls = p0 + p1;
      #pragma unroll
      for (int off = 1; off < 16; off <<= 1) ls += __shfl_xor(ls, off, 32);
      l_i[r] = l_i[r] * sc + ls;
      m_i[r] = mnew;
      pscale[r] = sc;
    }
    #pragma unroll
    for (int n = 0; n < 4; ++n)
      #pragma unroll
      for (int r = 0; r < 8; ++r) o[n][r] *= pscale[r];

    // P -> LDS (per-wave region), then A-fragment reload
    __bf16* pw = sP[w];
    #pragma unroll
    for (int nt = 0; nt < 2; ++nt)
      #pragma unroll
      for (int r = 0; r < 8; ++r)
        pw[(M0 + r) * 32 + nt * 16 + nc] = (__bf16)s[nt][r];

    // O += P V  (contraction over 32 keys)
    {
      const bf16x16 fa = load_frag_a(pw, 32, lane);
      #pragma unroll
      for (int n = 0; n < 4; ++n) {
        const bf16x16 fb = load_frag_b(&sVT[cur][(n * 16) * 32], 32, lane);
        o[n] = __builtin_amdgcn_wmma_f32_16x16x32_bf16(
            false, fa, false, fb, (short)0, o[n], false, false);
      }
    }

    wait_async0();        // next tile landed (overlapped with compute above)
    __syncthreads();
    cur ^= 1;
  }

  // epilogue: O / l  -> attn_out (96,1024,64) f32
  float inv[8];
  #pragma unroll
  for (int r = 0; r < 8; ++r) inv[r] = 1.f / l_i[r];
  #pragma unroll
  for (int n = 0; n < 4; ++n)
    #pragma unroll
    for (int r = 0; r < 8; ++r) {
      const int qrow = qbase + w * 16 + M0 + r;
      const int d = n * 16 + nc;
      ob[((size_t)b * HWDIM + qrow) * HD + d] = o[n][r] * inv[r];
    }
}

// ------------------------------------------------------------------
// Kernel 4: out = gather(attn_out)(8192x768) @ w_proj(768x768) + b_proj
// ------------------------------------------------------------------
__global__ __launch_bounds__(256)
void proj_gemm_kernel(const float* __restrict__ ao, const float* __restrict__ wp,
                      const float* __restrict__ bp, float* __restrict__ out)
{
  __shared__ __align__(16) __bf16 sA[128 * 32];
  __shared__ __align__(16) __bf16 sB[128 * 32];
  const int mbase = blockIdx.y * 128;
  const int nbase = blockIdx.x * 128;
  const int t = threadIdx.x;
  const int lane = t & 31, w = t >> 5;
  const int warpM = w & 3, warpN = w >> 2;

  const f32x8 fzero = {};
  f32x8 acc[2][4];
  #pragma unroll
  for (int i = 0; i < 2; ++i)
    #pragma unroll
    for (int j = 0; j < 4; ++j) acc[i][j] = fzero;

  for (int kk = 0; kk < CDIM; kk += 32) {
    // ---- batch loads (A has head-major gather: col k -> head=k>>6, d=k&63) ----
    float4 va[4], vbw[4];
    #pragma unroll
    for (int j = 0; j < 4; ++j) {
      const int f = t + 256 * j;
      const int row = f >> 3, c4 = (f & 7) << 2;
      const int mm = mbase + row;
      const int nimg = mm >> 10, hw = mm & 1023;
      const int cidx = kk + c4;
      const int head = cidx >> 6, dch = cidx & 63;
      va[j] = *reinterpret_cast<const float4*>(
          &ao[(((size_t)(nimg * NHEADS + head)) * HWDIM + hw) * HD + dch]);
    }
    #pragma unroll
    for (int j = 0; j < 4; ++j) {
      const int f = t + 256 * j;
      const int k = f >> 5, n4 = (f & 31) << 2;
      vbw[j] = *reinterpret_cast<const float4*>(
          &wp[(size_t)(kk + k) * CDIM + nbase + n4]);
    }
    // ---- convert + store ----
    #pragma unroll
    for (int j = 0; j < 4; ++j) {
      const int f = t + 256 * j;
      const int row = f >> 3, c4 = (f & 7) << 2;
      __bf16* d = &sA[row * 32 + c4];
      d[0] = (__bf16)va[j].x; d[1] = (__bf16)va[j].y;
      d[2] = (__bf16)va[j].z; d[3] = (__bf16)va[j].w;
    }
    #pragma unroll
    for (int j = 0; j < 4; ++j) {
      const int f = t + 256 * j;
      const int k = f >> 5, n4 = (f & 31) << 2;
      sB[(n4 + 0) * 32 + k] = (__bf16)vbw[j].x;
      sB[(n4 + 1) * 32 + k] = (__bf16)vbw[j].y;
      sB[(n4 + 2) * 32 + k] = (__bf16)vbw[j].z;
      sB[(n4 + 3) * 32 + k] = (__bf16)vbw[j].w;
    }
    if (kk + 32 < CDIM)
      __builtin_prefetch(&wp[(size_t)(kk + 32 + (t >> 3)) * CDIM + nbase], 0, 3);
    __syncthreads();

    bf16x16 fa[2], fb[4];
    #pragma unroll
    for (int mt = 0; mt < 2; ++mt)
      fa[mt] = load_frag_a(&sA[(warpM * 32 + mt * 16) * 32], 32, lane);
    #pragma unroll
    for (int nt = 0; nt < 4; ++nt)
      fb[nt] = load_frag_b(&sB[(warpN * 64 + nt * 16) * 32], 32, lane);
    #pragma unroll
    for (int mt = 0; mt < 2; ++mt)
      #pragma unroll
      for (int nt = 0; nt < 4; ++nt)
        acc[mt][nt] = __builtin_amdgcn_wmma_f32_16x16x32_bf16(
            false, fa[mt], false, fb[nt], (short)0, acc[mt][nt], false, false);
    __syncthreads();
  }

  const int M0 = (lane >> 4) << 3;
  const int nc = lane & 15;
  #pragma unroll
  for (int mt = 0; mt < 2; ++mt)
    #pragma unroll
    for (int nt = 0; nt < 4; ++nt) {
      const int nn = nbase + warpN * 64 + nt * 16 + nc;
      const float bias = bp[nn];
      #pragma unroll
      for (int r = 0; r < 8; ++r) {
        const int mm = mbase + warpM * 32 + mt * 16 + M0 + r;
        out[(size_t)mm * CDIM + nn] = acc[mt][nt][r] + bias;
      }
    }
}

// ------------------------------------------------------------------
extern "C" void kernel_launch(void* const* d_in, const int* in_sizes, int n_in,
                              void* d_out, int out_size, void* d_ws, size_t ws_size,
                              hipStream_t stream) {
  (void)in_sizes; (void)n_in; (void)out_size; (void)ws_size;
  const float* x     = (const float*)d_in[0];
  const float* wqkv  = (const float*)d_in[1];
  const float* bqkv  = (const float*)d_in[2];
  const float* wproj = (const float*)d_in[3];
  const float* bproj = (const float*)d_in[4];
  const float* rph   = (const float*)d_in[5];
  const float* rpw   = (const float*)d_in[6];
  float* out = (float*)d_out;

  char* ws = (char*)d_ws;
  size_t off = 0;
  auto carve = [&](size_t bytes) -> void* {
    off = (off + 255) & ~(size_t)255;
    void* p = ws + off;
    off += bytes;
    return p;
  };
  const size_t qkv_elems = (size_t)BATT * HWDIM * HD;      // 6.29M
  __bf16* qb  = (__bf16*)carve(qkv_elems * 2);
  __bf16* kb  = (__bf16*)carve(qkv_elems * 2);
  __bf16* vbT = (__bf16*)carve(qkv_elems * 2);             // transposed (96,64,1024)
  float*  bhb = (float*)carve((size_t)BATT * HWDIM * 32 * 4);
  float*  bwb = (float*)carve((size_t)BATT * HWDIM * 32 * 4);
  float*  aob = (float*)carve(qkv_elems * 4);

  qkv_gemm_kernel<<<dim3(2304 / 128, MROWS / 128), 256, 0, stream>>>(
      x, wqkv, bqkv, qb, kb, vbT);
  relpos_bias_kernel<<<(2 * BATT * HWDIM * 32) / 256, 256, 0, stream>>>(
      qb, rph, rpw, bhb, bwb);
  attn_kernel<<<dim3(BATT, HWDIM / 128), 256, 0, stream>>>(
      qb, kb, vbT, bhb, bwb, aob);
  proj_gemm_kernel<<<dim3(CDIM / 128, MROWS / 128), 256, 0, stream>>>(
      aob, wproj, bproj, out);
}
